// MyModel_61933428410185
// MI455X (gfx1250) — matrix operand order repaired and checked by reading, not compiled, not run on hardware
//
#include <hip/hip_runtime.h>

typedef __attribute__((ext_vector_type(16))) _Float16 v16h;
typedef __attribute__((ext_vector_type(8)))  _Float16 v8h;
typedef __attribute__((ext_vector_type(8)))  float    v8f;
typedef __attribute__((ext_vector_type(4)))  float    v4f;

#define NROWS  16384
#define DDIM   256
#define KTOP   5
#define BPITCH 264   // f16 per LDS B row: 256 + 8 pad -> 528B pitch, conflict-free b128 feeds
#define TPITCH 20    // f32 per spill-tile row: 16 + 4 pad -> conflict-free quad reads/stores

// Hard scheduling fence: nothing may be reordered across this point.
#if defined(__has_builtin) && __has_builtin(__builtin_amdgcn_sched_barrier)
#define SCHED_FENCE() __builtin_amdgcn_sched_barrier(0)
#else
#define SCHED_FENCE() asm volatile("" ::: "memory")
#endif

// Insert (val, index) into a register-resident, descending-sorted top-5 list.
#define TOP5_INSERT(tvArr, tiArr, val, index)                                   \
  do {                                                                          \
    if ((val) > tvArr[4]) {                                                     \
      tvArr[4] = (val); tiArr[4] = (index);                                     \
      _Pragma("unroll")                                                         \
      for (int _j = 4; _j > 0; --_j) {                                          \
        if (tvArr[_j] > tvArr[_j - 1]) {                                        \
          float _tv = tvArr[_j]; tvArr[_j] = tvArr[_j - 1]; tvArr[_j - 1] = _tv;\
          int   _ti = tiArr[_j]; tiArr[_j] = tiArr[_j - 1]; tiArr[_j - 1] = _ti;\
        }                                                                       \
      }                                                                         \
    }                                                                           \
  } while (0)

// ---------------------------------------------------------------------------
// Kernel 1: L2-normalize rows, store as f16 (WMMA input precision).
// ---------------------------------------------------------------------------
__global__ __launch_bounds__(256) void l2norm_f16_kernel(
    const float* __restrict__ feat, _Float16* __restrict__ nrm) {
  const int lane = threadIdx.x & 31;
  const int wave = threadIdx.x >> 5;
  const int row  = blockIdx.x * 8 + wave;

  const float* r = feat + (size_t)row * DDIM;
  float vals[8];
  float s = 0.0f;
#pragma unroll
  for (int i = 0; i < 8; ++i) {
    vals[i] = r[lane + i * 32];
    s += vals[i] * vals[i];
  }
#pragma unroll
  for (int off = 16; off > 0; off >>= 1)
    s += __shfl_xor(s, off, 32);
  const float inv = rsqrtf(s);

  _Float16* w = nrm + (size_t)row * DDIM;
#pragma unroll
  for (int i = 0; i < 8; ++i)
    w[lane + i * 32] = (_Float16)(vals[i] * inv);
}

// ---------------------------------------------------------------------------
// Kernel 2: fused  sim = norm @ norm.T  ->  streaming top-5  ->  weighted mean.
//
// Block = 128-row strip (8 waves x 16 rows). All waves share one 16-column
// B tile (16 x 256 f16 = 8 KB) staged in LDS by async-to-LDS copies with
// double buffering (ASYNCcnt). Per tile: issue ALL 16 ds_load_b128 B feeds,
// sched_barrier, then 8x v_wmma_f32_16x16x32_f16 with partial dscnt waits;
// per-wave padded LDS spill + s_wait_dscnt 0; register-resident top-5.
// ---------------------------------------------------------------------------
__global__ __launch_bounds__(256) void sim_topk_fused_kernel(
    const float* __restrict__ feat, const _Float16* __restrict__ nrm,
    float* __restrict__ out) {
  __shared__ _Float16 bbuf[2][16 * BPITCH];     // double-buffered B tile, 16.5 KB
  __shared__ float    tileLds[8][16][TPITCH];   // per-wave sim-tile spill, 10 KB
  __shared__ float    finW[128][KTOP];
  __shared__ int      finI[128][KTOP];

  const int tid  = threadIdx.x;
  const int wave = tid >> 5;
  const int lane = tid & 31;
  const int m    = lane & 15;       // row-in-tile (A) / col-in-tile (B)
  const int hi   = lane >> 4;       // K-half selector in this code's contiguous k order (k = 16*hi + s)
  const int rowBase = blockIdx.x * 128;

  // ---- Preload this wave's A-strip (16 rows x K=256) in WMMA A layout.
  v16h a[8];
  {
    const _Float16* arow = nrm + (size_t)(rowBase + wave * 16 + m) * DDIM;
#pragma unroll
    for (int kc = 0; kc < 8; ++kc) {
      const _Float16* p = arow + kc * 32 + hi * 16;
      v8h lo = *(const v8h*)(p);
      v8h hh = *(const v8h*)(p + 8);
#pragma unroll
      for (int j = 0; j < 8; ++j) { a[kc][j] = lo[j]; a[kc][8 + j] = hh[j]; }
    }
  }

  // ---- Async B-tile stage: thread t copies 32 B of the 8 KB tile.
  const unsigned bbufBase = (unsigned)(uintptr_t)(void*)&bbuf[0][0]; // LDS offset
  const int crow = tid >> 4;          // 0..15 : column-row within tile
  const int celt = (tid & 15) * 16;   // f16 element offset within row

  auto issueTile = [&](int tile, int buf) {
    const unsigned ldsOff =
        bbufBase + (unsigned)buf * (16 * BPITCH * 2) +
        (unsigned)crow * (BPITCH * 2) + (unsigned)celt * 2;
    const _Float16* g = nrm + (size_t)(tile * 16 + crow) * DDIM + celt;
    asm volatile("global_load_async_to_lds_b128 %0, %1, off"
                 :: "v"(ldsOff), "v"((unsigned long long)(uintptr_t)g)
                 : "memory");
    asm volatile("global_load_async_to_lds_b128 %0, %1, off"
                 :: "v"(ldsOff + 16u), "v"((unsigned long long)(uintptr_t)(g + 8))
                 : "memory");
  };

  float tv[KTOP]; int ti[KTOP];
#pragma unroll
  for (int k = 0; k < KTOP; ++k) { tv[k] = -1e30f; ti[k] = 0; }

  issueTile(0, 0);                       // prologue prefetch

  const int PASSES = NROWS / 16;         // 1024 column tiles
  for (int p = 0; p < PASSES; ++p) {
    const int cur = p & 1;

    if (p + 1 < PASSES) {
      issueTile(p + 1, cur ^ 1);         // prefetch next tile into other buffer
      // 2 batches in flight (2 async instrs each); <=2 left => tile p done.
      asm volatile("s_wait_asynccnt 2" ::: "memory");
    } else {
      asm volatile("s_wait_asynccnt 0" ::: "memory");
    }
    __syncthreads();                     // tile p visible to all waves

    // ---- Issue ALL 16 B-feed ds_load_b128s first; the sched fence stops the
    // scheduler from re-sinking them, so the waitcnt pass emits partial
    // s_wait_dscnt values and the 8 WMMAs pipeline against LDS latency.
    const _Float16* brow = &bbuf[cur][m * BPITCH + hi * 16];
    v16h b[8];
#pragma unroll
    for (int kc = 0; kc < 8; ++kc)
      b[kc] = *(const v16h*)(brow + kc * 32);

    SCHED_FENCE();

    v8f acc = {};
#pragma unroll
    for (int kc = 0; kc < 8; ++kc)
      acc = __builtin_amdgcn_wmma_f32_16x16x32_f16(
          false, a[kc], false, b[kc], (short)0, acc, false, false);

    // ---- Spill tile to this wave's private padded LDS region (C/D layout:
    // elem r -> row r + hi*8, col m). Pitch 20 => conflict-free.
#pragma unroll
    for (int r = 0; r < 8; ++r)
      tileLds[wave][r + hi * 8][m] = acc[r];

    // Wave-local ordering only: LDS is in-order per wave.
    asm volatile("s_wait_dscnt 0" ::: "memory");

    // ---- Lane merges 8 candidates of row m, cols hi*8..hi*8+7: two 16-B
    // quad reads (conflict-free), threshold-gated register insertion.
    const v4f q0 = *(const v4f*)&tileLds[wave][m][hi * 8];
    const v4f q1 = *(const v4f*)&tileLds[wave][m][hi * 8 + 4];
    float cv[8];
#pragma unroll
    for (int c = 0; c < 4; ++c) { cv[c] = q0[c]; cv[4 + c] = q1[c]; }
#pragma unroll
    for (int c = 0; c < 8; ++c) {
      if (cv[c] > tv[KTOP - 1]) {
        const int idx = p * 16 + hi * 8 + c;
        TOP5_INSERT(tv, ti, cv[c], idx);
      }
    }

    __syncthreads();   // all waves done reading bbuf[cur] before it is refilled
  }

  // ---- Merge the two half-lists of each row via lane<->lane^16 shuffles.
  float ov[KTOP]; int oi[KTOP];
#pragma unroll
  for (int k = 0; k < KTOP; ++k) {
    ov[k] = __shfl(tv[k], lane ^ 16, 32);
    oi[k] = __shfl(ti[k], lane ^ 16, 32);
  }
  if (lane < 16) {
#pragma unroll
    for (int k = 0; k < KTOP; ++k)
      if (ov[k] > tv[KTOP - 1]) TOP5_INSERT(tv, ti, ov[k], oi[k]);
    const int r = wave * 16 + m;
#pragma unroll
    for (int k = 0; k < KTOP; ++k) {
      finW[r][k] = tv[k] * tv[k];        // weights = sim^2
      finI[r][k] = ti[k];
    }
  }
  __syncthreads();

  // ---- Output: out[row] = 0.2 * sum_k w_k * feat[idx_k]; thread = dim d.
  // Indices/weights are uniform -> scalar loads; feat reads fully coalesced.
  const int d = tid;
  for (int r = 0; r < 128; ++r) {
    float s = 0.0f;
#pragma unroll
    for (int k = 0; k < KTOP; ++k)
      s += finW[r][k] * feat[(size_t)finI[r][k] * DDIM + d];
    out[(size_t)(rowBase + r) * DDIM + d] = s * 0.2f;
  }
}

// ---------------------------------------------------------------------------
extern "C" void kernel_launch(void* const* d_in, const int* in_sizes, int n_in,
                              void* d_out, int out_size, void* d_ws, size_t ws_size,
                              hipStream_t stream) {
  const float* feat = (const float*)d_in[0];
  float*       out  = (float*)d_out;
  _Float16*    nrm  = (_Float16*)d_ws;      // 16384*256*2 = 8 MB scratch

  const int n = in_sizes[0] / DDIM;         // 16384

  l2norm_f16_kernel<<<n / 8, 256, 0, stream>>>(feat, nrm);
  sim_topk_fused_kernel<<<n / 128, 256, 0, stream>>>(feat, nrm, out);
}
